// MeshConv_9216999817472
// MI455X (gfx1250) — matrix operand order repaired and checked by reading, not compiled, not run on hardware
//
#include <hip/hip_runtime.h>
#include <hip/hip_bf16.h>

// Problem constants (fixed by setup_inputs)
#define BATCH 8
#define CIN   32
#define NE    100000
#define COUT  64
#define KW    5
#define CK    160              // CIN*KW
#define TPB_  6250             // 16-edge tiles per batch: NE/16
#define GSTR  172              // padded G row stride (floats): 44 mod 64 banks
#define WSTR  168              // padded W row stride (bf16) = 84 dwords = 21 uint4

typedef __attribute__((ext_vector_type(16))) __bf16 v16bf;
typedef __attribute__((ext_vector_type(8)))  float  v8f;
typedef unsigned int u32;

union frag_u { v16bf v; uint4 q[2]; };
union accu_t { v8f v; float4 q[2]; };

// ---------------------------------------------------------------------------
// Kernel 1: zero the pad row (index 0) of xt for each batch
// ---------------------------------------------------------------------------
__global__ void meshconv_zeropad(float* __restrict__ xt) {
    int t = threadIdx.x;              // 256 threads: 8 batches * 32 channels
    int b = t >> 5, c = t & 31;
    xt[((size_t)b * (NE + 1) + 0) * CIN + c] = 0.0f;
}

// ---------------------------------------------------------------------------
// Kernel 2: transpose x (B,C,E) -> xt (B,E+1,C), rows shifted by +1
// ---------------------------------------------------------------------------
__global__ void meshconv_transpose(const float* __restrict__ x,
                                   float* __restrict__ xt) {
    __shared__ float tile[CIN][65];
    const int tilesPerB = (NE + 63) / 64;           // 1563
    int b  = blockIdx.x / tilesPerB;
    int e0 = (blockIdx.x % tilesPerB) * 64;
    int t  = threadIdx.x;

    int el = t & 63, qq = t >> 6;                   // read: coalesced along E
#pragma unroll
    for (int i = 0; i < 8; ++i) {
        int c = qq * 8 + i;
        int e = e0 + el;
        tile[c][el] = (e < NE) ? x[((size_t)b * CIN + c) * NE + e] : 0.0f;
    }
    __syncthreads();

    int c = t & 31, rg = t >> 5;                    // write: coalesced along C
#pragma unroll
    for (int i = 0; i < 8; ++i) {
        int el2 = rg * 8 + i;
        int e = e0 + el2;
        if (e < NE)
            xt[((size_t)b * (NE + 1) + e + 1) * CIN + c] = tile[c][el2];
    }
}

// ---------------------------------------------------------------------------
// Kernel 3: one-time bf16 hi/lo split of the constant weight matrix
// ---------------------------------------------------------------------------
__global__ void meshconv_wsplit(const float* __restrict__ W,
                                __bf16* __restrict__ whi,
                                __bf16* __restrict__ wlo) {
    int i = blockIdx.x * 256 + threadIdx.x;
    if (i < COUT * CK) {
        float w  = W[i];
        __bf16 h = (__bf16)w;
        whi[i] = h;
        wlo[i] = (__bf16)(w - (float)h);
    }
}

// ---------------------------------------------------------------------------
// bf16 hi/lo split of 16 fp32 values into two WMMA fragments (B side only)
// ---------------------------------------------------------------------------
__device__ __forceinline__ void split_bf16(const float* __restrict__ t,
                                           v16bf& hi, v16bf& lo) {
#pragma unroll
    for (int i = 0; i < 16; ++i) {
        __bf16 h = (__bf16)t[i];
        hi[i] = h;
        lo[i] = (__bf16)(t[i] - (float)h);
    }
}

// ---------------------------------------------------------------------------
// Kernel 4: gather + symmetric features + WMMA GEMM (+bias) per 16-edge tile
//   Dynamic LDS layout (bytes):
//     whi_s : [64][WSTR] bf16   21504
//     wlo_s : [64][WSTR] bf16   21504
//     b_s   : [64] f32            256
//     g_s   : 8 * [16][GSTR] f32 88064   -> total 131328 B
// ---------------------------------------------------------------------------
__global__ void meshconv_main(const float*  __restrict__ xt,
                              const int*    __restrict__ ge,
                              const __bf16* __restrict__ whi,
                              const __bf16* __restrict__ wlo,
                              const float*  __restrict__ bias,
                              float* __restrict__ out) {
    extern __shared__ float smem[];
    __bf16* whi_s = (__bf16*)smem;                       // 64*WSTR bf16
    __bf16* wlo_s = whi_s + COUT * WSTR;                 // 64*WSTR bf16
    float*  b_s   = (float*)(wlo_s + COUT * WSTR);       // 64 f32
    float*  g_s   = b_s + 64;                            // 8*16*GSTR f32

    const int tid  = threadIdx.x;
    const int lane = tid & 31;
    const int wave = tid >> 5;
    const int col  = lane & 15;              // edge-within-tile / N column
    const int half = lane >> 4;              // 0: lanes 0-15, 1: lanes 16-31

    // ---- stage pre-split weights (b128-packed) & bias into LDS ----
    {
        const uint4* src_h = (const uint4*)whi;   // 1280 uint4 per array
        const uint4* src_l = (const uint4*)wlo;
        uint4* dst_h = (uint4*)whi_s;             // LDS row = 21 uint4 (84 dw)
        uint4* dst_l = (uint4*)wlo_s;
#pragma unroll
        for (int i = tid; i < COUT * 20; i += 256) {   // 5 iterations
            int row = i / 20, c = i % 20;
            dst_h[row * 21 + c] = src_h[i];
            dst_l[row * 21 + c] = src_l[i];
        }
    }
    if (tid < COUT) b_s[tid] = bias[tid];

    // ---- this wave's tile ----
    const int t  = blockIdx.x * 8 + wave;    // 0..49999
    const int b  = t / TPB_;
    const int e0 = (t % TPB_) * 16;
    const int e  = e0 + col;

    // ---- gather 5 neighbor rows, build symmetric features into g_s ----
    int idx[5];
    idx[0] = e + 1;                          // self (+1 for pad row)
    {
        int4 gi = *(const int4*)(ge + ((size_t)b * NE + e) * 4);
        idx[1] = gi.x + 1; idx[2] = gi.y + 1;
        idx[3] = gi.z + 1; idx[4] = gi.w + 1;
    }
    const float* xb   = xt + (size_t)b * (NE + 1) * CIN;
    float*       grow = g_s + wave * (16 * GSTR) + col * GSTR;

#pragma unroll
    for (int cc = 0; cc < 4; ++cc) {
        const int c0 = half * 16 + cc * 4;   // 4 channels per pass
        float fv[5][4];
#pragma unroll
        for (int s = 0; s < 5; ++s) {
            float4 v = *(const float4*)(xb + (size_t)idx[s] * CIN + c0);
            fv[s][0] = v.x; fv[s][1] = v.y; fv[s][2] = v.z; fv[s][3] = v.w;
        }
        float gv[20] __attribute__((aligned(16)));
#pragma unroll
        for (int j = 0; j < 4; ++j) {
            gv[j * 5 + 0] = fv[0][j];
            gv[j * 5 + 1] = fv[1][j] + fv[3][j];
            gv[j * 5 + 2] = fv[2][j] + fv[4][j];
            gv[j * 5 + 3] = fabsf(fv[1][j] - fv[3][j]);
            gv[j * 5 + 4] = fabsf(fv[2][j] - fv[4][j]);
        }
        float4* dst = (float4*)(grow + c0 * 5);     // c0*5 is 16B-aligned
#pragma unroll
        for (int i = 0; i < 5; ++i) dst[i] = ((const float4*)gv)[i];
    }

    __syncthreads();   // W/bias staged + all G rows of this wave visible

    // ---- accumulators pre-loaded with bias (row o = mt*16 + 8*half + r) ----
    accu_t acc[4];
#pragma unroll
    for (int mt = 0; mt < 4; ++mt) {
        const float* bp = b_s + mt * 16 + half * 8;
        acc[mt].q[0] = ((const float4*)bp)[0];
        acc[mt].q[1] = ((const float4*)bp)[1];
    }

    // ---- GEMM: out(64x16) = W(64x160) * G(160x16), 3-term bf16 split ----
    const float* grd = g_s + wave * (16 * GSTR) + col * GSTR + half * 16;

#pragma unroll
    for (int qk = 0; qk < 5; ++qk) {         // K chunks of 32
        // B fragment: lane holds K = qk*32 + half*16 + [0..15] of column `col`
        float gt[16] __attribute__((aligned(16)));
        {
            const float4* src = (const float4*)(grd + qk * 32);
#pragma unroll
            for (int i = 0; i < 4; ++i) ((float4*)gt)[i] = src[i];
        }
        v16bf bhi, blo;
        split_bf16(gt, bhi, blo);

#pragma unroll
        for (int mt = 0; mt < 4; ++mt) {
            // A fragment: row = mt*16+col; K = qk*32 + half*16 + [0..15]
            const int wo = (mt * 16 + col) * WSTR + qk * 32 + half * 16;
            frag_u ahi, alo;
            ahi.q[0] = *(const uint4*)(whi_s + wo);
            ahi.q[1] = *(const uint4*)(whi_s + wo + 8);
            alo.q[0] = *(const uint4*)(wlo_s + wo);
            alo.q[1] = *(const uint4*)(wlo_s + wo + 8);

            acc[mt].v = __builtin_amdgcn_wmma_f32_16x16x32_bf16(
                false, alo.v, false, bhi, (short)0, acc[mt].v, false, false);
            acc[mt].v = __builtin_amdgcn_wmma_f32_16x16x32_bf16(
                false, ahi.v, false, blo, (short)0, acc[mt].v, false, false);
            acc[mt].v = __builtin_amdgcn_wmma_f32_16x16x32_bf16(
                false, ahi.v, false, bhi, (short)0, acc[mt].v, false, false);
        }
    }

    // ---- epilogue: D vgpr r -> M = r + 8*half, N = col; store ----
#pragma unroll
    for (int mt = 0; mt < 4; ++mt) {
        float* op = out + ((size_t)b * COUT + mt * 16 + half * 8) * NE
                        + (e0 + col);
#pragma unroll
        for (int r = 0; r < 8; ++r)
            op[(size_t)r * NE] = acc[mt].v[r];
    }
}

// ---------------------------------------------------------------------------
extern "C" void kernel_launch(void* const* d_in, const int* in_sizes, int n_in,
                              void* d_out, int out_size, void* d_ws, size_t ws_size,
                              hipStream_t stream) {
    const float* x    = (const float*)d_in[0];   // (B, C, E, 1)
    const int*   ge   = (const int*)  d_in[1];   // (B, E, 4)
    const float* W    = (const float*)d_in[2];   // (COUT, CIN, 1, K)
    const float* bias = (const float*)d_in[3];   // (COUT,)
    float*       out  = (float*)d_out;           // (B, COUT, E, 1)

    // workspace carve-up
    float*  xt  = (float*)d_ws;                               // 102,401,024 B
    __bf16* whi = (__bf16*)((char*)d_ws + (size_t)BATCH * (NE + 1) * CIN * 4);
    __bf16* wlo = whi + COUT * CK;                            // +20,480 B each

    meshconv_zeropad<<<1, 256, 0, stream>>>(xt);

    const int tilesPerB = (NE + 63) / 64;        // 1563
    meshconv_transpose<<<BATCH * tilesPerB, 256, 0, stream>>>(x, xt);

    meshconv_wsplit<<<(COUT * CK + 255) / 256, 256, 0, stream>>>(W, whi, wlo);

    const size_t shbytes = (size_t)(2 * COUT * WSTR) * sizeof(__bf16)
                         + (size_t)(64 + 8 * 16 * GSTR) * sizeof(float);
    meshconv_main<<<(BATCH * TPB_) / 8, 256, shbytes, stream>>>(
        xt, ge, whi, wlo, bias, out);
}
